// Transformer_9285719294210
// MI455X (gfx1250) — matrix operand order repaired and checked by reading, not compiled, not run on hardware
//
#include <hip/hip_runtime.h>

#define Bb   2
#define Nn   4096
#define Dd   128
#define Hh   4
#define HDd  32
#define Ll   2
#define M8   (Bb * Nn)          // 8192 rows
#define BND  (Bb * Nn * Dd)     // 1048576 elements

typedef __attribute__((ext_vector_type(16))) _Float16 v16h;
typedef __attribute__((ext_vector_type(8)))  float    v8f;

// -------- helpers --------
__device__ inline v16h ldg_v16h(const _Float16* p0, const _Float16* p1) {
  union { v16h v; float4 f[2]; } u;
  u.f[0] = *reinterpret_cast<const float4*>(p0);
  u.f[1] = *reinterpret_cast<const float4*>(p1);
  return u.v;
}

// -------- BatchNorm: per-column mean / biased var --------
__global__ __launch_bounds__(256) void bn_stats_kernel(const float* __restrict__ x,
                                                       float* __restrict__ stats) {
  int c = blockIdx.x;           // column 0..Dd-1
  int t = threadIdx.x;
  float s = 0.f, sq = 0.f;
  for (int r = t; r < M8; r += 256) {
    float v = x[(size_t)r * Dd + c];
    s += v; sq += v * v;
  }
  __shared__ float rs[256], rq[256];
  rs[t] = s; rq[t] = sq;
  __syncthreads();
  for (int o = 128; o > 0; o >>= 1) {
    if (t < o) { rs[t] += rs[t + o]; rq[t] += rq[t + o]; }
    __syncthreads();
  }
  if (t == 0) {
    float mu  = rs[0] / (float)M8;
    float var = rq[0] / (float)M8 - mu * mu;
    stats[2 * c]     = mu;
    stats[2 * c + 1] = rsqrtf(var + 1e-5f);
  }
}

__global__ __launch_bounds__(256) void bn_apply_kernel(const float* __restrict__ x,
                                                       const float* __restrict__ stats,
                                                       const float* __restrict__ g,
                                                       const float* __restrict__ bta,
                                                       _Float16* __restrict__ out) {
  int i = blockIdx.x * 256 + threadIdx.x;
  if (i >= BND) return;
  int c = i & (Dd - 1);
  float v = (x[i] - stats[2 * c]) * stats[2 * c + 1] * g[c] + bta[c];
  out[i] = (_Float16)v;
}

// -------- weight convert + transpose: Wt[n*K+k] = (f16)W[k*Nout+n] --------
__global__ __launch_bounds__(256) void wconv_kernel(const float* __restrict__ w,
                                                    _Float16* __restrict__ wt,
                                                    int K, int Nout) {
  int i = blockIdx.x * 256 + threadIdx.x;
  if (i >= K * Nout) return;
  int n = i / K, kk = i - n * K;
  wt[i] = (_Float16)w[(size_t)kk * Nout + n];
}

// -------- scale + f32->f16 convert --------
__global__ __launch_bounds__(256) void sconv_kernel(const float* __restrict__ in,
                                                    _Float16* __restrict__ out,
                                                    int total, float scale) {
  int i = blockIdx.x * 256 + threadIdx.x;
  if (i >= total) return;
  out[i] = (_Float16)(in[i] * scale);
}

// -------- RoPE (2D positions, NF=32), fold in scale --------
__global__ __launch_bounds__(256) void rope_kernel(const float* __restrict__ in,
                                                   const float* __restrict__ pos,
                                                   _Float16* __restrict__ out,
                                                   float scale) {
  const int total = M8 * (Dd / 2);
  int i = blockIdx.x * 256 + threadIdx.x;
  if (i >= total) return;
  int pair = i & (Dd / 2 - 1);          // 0..63
  int bn   = i / (Dd / 2);              // 0..8191
  int sel  = pair >= 32 ? 1 : 0;        // pos dimension
  int f    = pair - sel * 32;           // frequency index 0..31
  float freq = 3.14159265358979323846f * (float)(f + 1);  // 2*pi*(f+1)/PERIOD, PERIOD=2
  float a = pos[bn * 2 + sel] * freq;
  float c = cosf(a), s = sinf(a);
  int d0 = pair * 2;
  float x0 = in[(size_t)bn * Dd + d0];
  float x1 = in[(size_t)bn * Dd + d0 + 1];
  out[(size_t)bn * Dd + d0]     = (_Float16)((c * x0 - s * x1) * scale);
  out[(size_t)bn * Dd + d0 + 1] = (_Float16)((c * x1 + s * x0) * scale);
}

// -------- V transpose: vt[(b*H+h)*HD+hd][n] = v[b][n][h*HD+hd] --------
__global__ __launch_bounds__(256) void vtrans_kernel(const float* __restrict__ v,
                                                     _Float16* __restrict__ vt) {
  int i = blockIdx.x * 256 + threadIdx.x;
  if (i >= BND) return;
  int b   = i / (Nn * Dd);
  int rem = i - b * (Nn * Dd);
  int n   = rem / Dd;
  int d   = rem - n * Dd;
  int h   = d / HDd, hd = d - h * HDd;
  vt[(((size_t)b * Hh + h) * HDd + hd) * Nn + n] = (_Float16)v[i];
}

// -------- generic WMMA GEMM: one wave -> 16x64 tile --------
// Y[M x Nout] = X[M x K] @ W[K x Nout] (W given transposed: Wt[Nout x K]) + bias
__global__ __launch_bounds__(32) void gemm_kernel(const _Float16* __restrict__ X,
                                                  const _Float16* __restrict__ Wt,
                                                  const float* __restrict__ bias,
                                                  const float* __restrict__ resid,
                                                  float* __restrict__ outf,
                                                  _Float16* __restrict__ outh,
                                                  int K, int Nout, int relu) {
  int lane = threadIdx.x;
  int lmod = lane & 15, lhi = lane >> 4;
  int mbase = blockIdx.x * 16;
  int nbase = blockIdx.y * 64;
  int arow  = mbase + lmod;

  v8f acc0 = {0,0,0,0,0,0,0,0};
  v8f acc1 = acc0, acc2 = acc0, acc3 = acc0;
  const _Float16* xr = X + (size_t)arow * K;

  for (int kc = 0; kc < K; kc += 32) {
    v16h a = ldg_v16h(xr + kc + lhi * 16, xr + kc + lhi * 16 + 8);
    const _Float16* w0 = Wt + (size_t)(nbase +      lmod) * K + kc + lhi * 16;
    const _Float16* w1 = Wt + (size_t)(nbase + 16 + lmod) * K + kc + lhi * 16;
    const _Float16* w2 = Wt + (size_t)(nbase + 32 + lmod) * K + kc + lhi * 16;
    const _Float16* w3 = Wt + (size_t)(nbase + 48 + lmod) * K + kc + lhi * 16;
    v16h b0 = ldg_v16h(w0, w0 + 8);
    v16h b1 = ldg_v16h(w1, w1 + 8);
    v16h b2 = ldg_v16h(w2, w2 + 8);
    v16h b3 = ldg_v16h(w3, w3 + 8);
    acc0 = __builtin_amdgcn_wmma_f32_16x16x32_f16(false, a, false, b0, (short)0, acc0, false, false);
    acc1 = __builtin_amdgcn_wmma_f32_16x16x32_f16(false, a, false, b1, (short)0, acc1, false, false);
    acc2 = __builtin_amdgcn_wmma_f32_16x16x32_f16(false, a, false, b2, (short)0, acc2, false, false);
    acc3 = __builtin_amdgcn_wmma_f32_16x16x32_f16(false, a, false, b3, (short)0, acc3, false, false);
  }

  v8f accs[4] = {acc0, acc1, acc2, acc3};
#pragma unroll
  for (int t = 0; t < 4; t++) {
#pragma unroll
    for (int r = 0; r < 8; r++) {
      int col  = nbase + t * 16 + lmod;
      int orow = mbase + r + 8 * lhi;
      float v = accs[t][r] + bias[col];
      if (relu) v = fmaxf(v, 0.f);
      size_t idx = (size_t)orow * Nout + col;
      if (resid) v += resid[idx];
      if (outf)  outf[idx] = v;
      if (outh)  outh[idx] = (_Float16)v;
    }
  }
}

// -------- flash attention, windowed mask, one wave = 16 query rows of one (b,h) --------
__global__ __launch_bounds__(32) void attn_kernel(const _Float16* __restrict__ q,
                                                  const _Float16* __restrict__ k,
                                                  const _Float16* __restrict__ vt,
                                                  const float* __restrict__ pos,
                                                  _Float16* __restrict__ o) {
  int wid = blockIdx.x;                 // B*H*(N/16)
  int qt  = wid & (Nn / 16 - 1);
  int bh  = wid / (Nn / 16);
  int h   = bh & (Hh - 1), b = bh / Hh;
  int lane = threadIdx.x;
  int lmod = lane & 15, lhi = lane >> 4;
  int qbase = qt * 16;

  const _Float16* qb  = q  + (size_t)b * Nn * Dd + h * HDd;
  const _Float16* kb_ = k  + (size_t)b * Nn * Dd + h * HDd;
  const _Float16* vtp = vt + ((size_t)(b * Hh + h) * HDd) * Nn;

  // A operand: 16x32 query tile (fixed across the key loop)
  int arow = qbase + lmod;
  v16h A = ldg_v16h(qb + (size_t)arow * Dd + lhi * 16,
                    qb + (size_t)arow * Dd + lhi * 16 + 8);

  // per-lane query positions for the 8 rows this lane owns in C-layout
  float qpx[8], qpy[8];
#pragma unroll
  for (int r = 0; r < 8; r++) {
    int qr = qbase + r + 8 * lhi;
    qpx[r] = pos[((size_t)b * Nn + qr) * 2];
    qpy[r] = pos[((size_t)b * Nn + qr) * 2 + 1];
  }

  v8f O0 = {0,0,0,0,0,0,0,0};
  v8f O1 = O0;
  v8f zc = O0;
  float mi[8], li[8];
#pragma unroll
  for (int r = 0; r < 8; r++) { mi[r] = -1e30f; li[r] = 0.f; }

  __shared__ _Float16 lds[16 * 32];     // probs tile, [row m][key kk] row stride 32

  for (int kbase = 0; kbase < Nn; kbase += 32) {
    // B operands for scores: 16 key rows each (contiguous head-dim reads)
    const _Float16* kr0 = kb_ + (size_t)(kbase +      lmod) * Dd + lhi * 16;
    const _Float16* kr1 = kb_ + (size_t)(kbase + 16 + lmod) * Dd + lhi * 16;
    v16h B0 = ldg_v16h(kr0, kr0 + 8);
    v16h B1 = ldg_v16h(kr1, kr1 + 8);
    v8f S0 = __builtin_amdgcn_wmma_f32_16x16x32_f16(false, A, false, B0, (short)0, zc, false, false);
    v8f S1 = __builtin_amdgcn_wmma_f32_16x16x32_f16(false, A, false, B1, (short)0, zc, false, false);

    // key positions for this lane's columns
    float kx0 = pos[((size_t)b * Nn + kbase + lmod) * 2];
    float ky0 = pos[((size_t)b * Nn + kbase + lmod) * 2 + 1];
    float kx1 = pos[((size_t)b * Nn + kbase + 16 + lmod) * 2];
    float ky1 = pos[((size_t)b * Nn + kbase + 16 + lmod) * 2 + 1];

#pragma unroll
    for (int r = 0; r < 8; r++) {
      float dx0 = qpx[r] - kx0, dy0 = qpy[r] - ky0;
      float dx1 = qpx[r] - kx1, dy1 = qpy[r] - ky1;
      float s0 = (dx0 * dx0 + dy0 * dy0 <= 0.0625f) ? S0[r] : -1e30f;
      float s1 = (dx1 * dx1 + dy1 * dy1 <= 0.0625f) ? S1[r] : -1e30f;
      // row max over 32 keys: combine tiles, reduce across 16 lanes of this half
      float t = fmaxf(s0, s1);
      t = fmaxf(t, __shfl_xor(t, 1, 32));
      t = fmaxf(t, __shfl_xor(t, 2, 32));
      t = fmaxf(t, __shfl_xor(t, 4, 32));
      t = fmaxf(t, __shfl_xor(t, 8, 32));
      float m2 = fmaxf(mi[r], t);
      float al = __expf(mi[r] - m2);
      float p0 = __expf(s0 - m2);
      float p1 = __expf(s1 - m2);
      float rs = p0 + p1;
      rs += __shfl_xor(rs, 1, 32);
      rs += __shfl_xor(rs, 2, 32);
      rs += __shfl_xor(rs, 4, 32);
      rs += __shfl_xor(rs, 8, 32);
      li[r] = li[r] * al + rs;
      O0[r] *= al;
      O1[r] *= al;
      mi[r] = m2;
      // stage probs to LDS (C-layout -> [m][kk] f16)
      int m = r + 8 * lhi;
      lds[m * 32 + lmod]      = (_Float16)p0;
      lds[m * 32 + 16 + lmod] = (_Float16)p1;
    }

    asm volatile("s_wait_dscnt 0" ::: "memory");

    // re-read probs in A-layout
    v16h Ap = ldg_v16h(&lds[lmod * 32 + lhi * 16],
                       &lds[lmod * 32 + lhi * 16 + 8]);
    // B operands from V^T: column = output dim, contiguous keys
    const _Float16* v0 = vtp + (size_t)lmod * Nn        + kbase + lhi * 16;
    const _Float16* v1 = vtp + (size_t)(16 + lmod) * Nn + kbase + lhi * 16;
    v16h BV0 = ldg_v16h(v0, v0 + 8);
    v16h BV1 = ldg_v16h(v1, v1 + 8);
    O0 = __builtin_amdgcn_wmma_f32_16x16x32_f16(false, Ap, false, BV0, (short)0, O0, false, false);
    O1 = __builtin_amdgcn_wmma_f32_16x16x32_f16(false, Ap, false, BV1, (short)0, O1, false, false);
  }

  _Float16* ob = o + (size_t)b * Nn * Dd + h * HDd;
#pragma unroll
  for (int r = 0; r < 8; r++) {
    int qr = qbase + r + 8 * lhi;
    float inv = 1.f / li[r];
    ob[(size_t)qr * Dd + lmod]      = (_Float16)(O0[r] * inv);
    ob[(size_t)qr * Dd + 16 + lmod] = (_Float16)(O1[r] * inv);
  }
}

// ----------------------------- host orchestration -----------------------------
extern "C" void kernel_launch(void* const* d_in, const int* in_sizes, int n_in,
                              void* d_out, int out_size, void* d_ws, size_t ws_size,
                              hipStream_t stream) {
  const float* x_in  = (const float*)d_in[0];
  const float* pos   = (const float*)d_in[1];
  const float* Wq_w  = (const float*)d_in[2];
  const float* Wq_b  = (const float*)d_in[3];
  const float* Wk_w  = (const float*)d_in[4];
  const float* Wk_b  = (const float*)d_in[5];
  const float* Wv_w  = (const float*)d_in[6];
  const float* Wv_b  = (const float*)d_in[7];
  const float* Wo_w  = (const float*)d_in[8];
  const float* Wo_b  = (const float*)d_in[9];
  const float* n1_g  = (const float*)d_in[10];
  const float* n1_b  = (const float*)d_in[11];
  const float* n2_g  = (const float*)d_in[12];
  const float* n2_b  = (const float*)d_in[13];
  const float* l1_w  = (const float*)d_in[14];
  const float* l1_b  = (const float*)d_in[15];
  const float* l2_w  = (const float*)d_in[16];
  const float* l2_b  = (const float*)d_in[17];

  unsigned char* ws = (unsigned char*)d_ws;
  size_t off = 0;
  auto carve = [&](size_t bytes) -> void* {
    void* p = ws + off;
    off = (off + bytes + 255) & ~(size_t)255;
    return p;
  };
  float*    xbuf  = (float*)   carve((size_t)BND * 4);
  float*    qf    = (float*)   carve((size_t)BND * 4);
  float*    kf    = (float*)   carve((size_t)BND * 4);
  float*    vf    = (float*)   carve((size_t)BND * 4);
  float*    stats = (float*)   carve(2 * 256 * 4);
  _Float16* xnh   = (_Float16*)carve((size_t)BND * 2);
  _Float16* qh    = (_Float16*)carve((size_t)BND * 2);
  _Float16* kh    = (_Float16*)carve((size_t)BND * 2);
  _Float16* vth   = (_Float16*)carve((size_t)BND * 2);
  _Float16* oh    = (_Float16*)carve((size_t)BND * 2);
  _Float16* hh    = (_Float16*)carve((size_t)Bb * Nn * 2 * Dd * 2);
  _Float16* wt    = (_Float16*)carve((size_t)2 * Dd * Dd * 2);

  const float scale = 0.17677669529663687f;  // 1/sqrt(HD)
  const int eltBlocks = (BND + 255) / 256;

  hipMemcpyAsync(xbuf, x_in, (size_t)BND * 4, hipMemcpyDeviceToDevice, stream);

  for (int i = 0; i < Ll; i++) {
    // ---- pre-norm 1 ----
    bn_stats_kernel<<<Dd, 256, 0, stream>>>(xbuf, stats);
    bn_apply_kernel<<<eltBlocks, 256, 0, stream>>>(xbuf, stats, n1_g + i * Dd, n1_b + i * Dd, xnh);

    // ---- Q, K, V projections ----
    wconv_kernel<<<(Dd * Dd + 255) / 256, 256, 0, stream>>>(Wq_w + (size_t)i * Dd * Dd, wt, Dd, Dd);
    gemm_kernel<<<dim3(M8 / 16, Dd / 64), 32, 0, stream>>>(xnh, wt, Wq_b + i * Dd, nullptr, qf, nullptr, Dd, Dd, 0);
    wconv_kernel<<<(Dd * Dd + 255) / 256, 256, 0, stream>>>(Wk_w + (size_t)i * Dd * Dd, wt, Dd, Dd);
    gemm_kernel<<<dim3(M8 / 16, Dd / 64), 32, 0, stream>>>(xnh, wt, Wk_b + i * Dd, nullptr, kf, nullptr, Dd, Dd, 0);
    wconv_kernel<<<(Dd * Dd + 255) / 256, 256, 0, stream>>>(Wv_w + (size_t)i * Dd * Dd, wt, Dd, Dd);
    gemm_kernel<<<dim3(M8 / 16, Dd / 64), 32, 0, stream>>>(xnh, wt, Wv_b + i * Dd, nullptr, vf, nullptr, Dd, Dd, 0);

    // ---- RoPE (layer 0 only) + f16 conversion, fold score scale into q ----
    if (i == 0) {
      rope_kernel<<<(M8 * (Dd / 2) + 255) / 256, 256, 0, stream>>>(qf, pos, qh, scale);
      rope_kernel<<<(M8 * (Dd / 2) + 255) / 256, 256, 0, stream>>>(kf, pos, kh, 1.0f);
    } else {
      sconv_kernel<<<eltBlocks, 256, 0, stream>>>(qf, qh, BND, scale);
      sconv_kernel<<<eltBlocks, 256, 0, stream>>>(kf, kh, BND, 1.0f);
    }
    vtrans_kernel<<<eltBlocks, 256, 0, stream>>>(vf, vth);

    // ---- windowed flash attention ----
    attn_kernel<<<Bb * Hh * (Nn / 16), 32, 0, stream>>>(qh, kh, vth, pos, oh);

    // ---- output projection + residual ----
    wconv_kernel<<<(Dd * Dd + 255) / 256, 256, 0, stream>>>(Wo_w + (size_t)i * Dd * Dd, wt, Dd, Dd);
    gemm_kernel<<<dim3(M8 / 16, Dd / 64), 32, 0, stream>>>(oh, wt, Wo_b + i * Dd, xbuf, xbuf, nullptr, Dd, Dd, 0);

    // ---- pre-norm 2 + FFN + residual ----
    bn_stats_kernel<<<Dd, 256, 0, stream>>>(xbuf, stats);
    bn_apply_kernel<<<eltBlocks, 256, 0, stream>>>(xbuf, stats, n2_g + i * Dd, n2_b + i * Dd, xnh);
    wconv_kernel<<<(Dd * 2 * Dd + 255) / 256, 256, 0, stream>>>(l1_w + (size_t)i * Dd * 2 * Dd, wt, Dd, 2 * Dd);
    gemm_kernel<<<dim3(M8 / 16, (2 * Dd) / 64), 32, 0, stream>>>(xnh, wt, l1_b + i * 2 * Dd, nullptr, nullptr, hh, Dd, 2 * Dd, 1);
    wconv_kernel<<<(2 * Dd * Dd + 255) / 256, 256, 0, stream>>>(l2_w + (size_t)i * 2 * Dd * Dd, wt, 2 * Dd, Dd);
    gemm_kernel<<<dim3(M8 / 16, Dd / 64), 32, 0, stream>>>(hh, wt, l2_b + i * Dd, xbuf, xbuf, nullptr, 2 * Dd, Dd, 0);
  }

  hipMemcpyAsync(d_out, xbuf, (size_t)BND * 4, hipMemcpyDeviceToDevice, stream);
}
